// QuantumLLM_46608985096301
// MI455X (gfx1250) — matrix operand order repaired and checked by reading, not compiled, not run on hardware
//
#include <hip/hip_runtime.h>
#include <cmath>

#define DIM 256
#define VOCAB 16384
#define BB 2
#define SS 4096
#define NROWS (BB*SS)
#define NLAYERS 3
#define QTILE 64
#define KTILE 32

typedef unsigned short ushort_t;
typedef __attribute__((ext_vector_type(16))) __bf16 v16bf;
typedef __attribute__((ext_vector_type(8)))  float  v8f;

union BF16Frag { uint4 q[2]; v16bf v; };

__device__ __forceinline__ ushort_t f2bf(float f){
  unsigned int u = __float_as_uint(f);
  u += 0x7FFFu + ((u >> 16) & 1u);          // round-to-nearest-even
  return (ushort_t)(u >> 16);
}
__device__ __forceinline__ float clampf(float v, float lo, float hi){
  return fminf(fmaxf(v, lo), hi);
}
__device__ __forceinline__ v8f wmma_bf16(v16bf a, v16bf b, v8f c){
  return __builtin_amdgcn_wmma_f32_16x16x32_bf16(false, a, false, b, (short)0, c, false, false);
}
// A-fragment (16x32 bf16): row = row0+laneLo; K = hf*16 + j
__device__ __forceinline__ v16bf loadFragA(const ushort_t* p, int hf){
  BF16Frag f;
  f.q[0] = *(const uint4*)(p + hf*16);
  f.q[1] = *(const uint4*)(p + hf*16 + 8);
  return f.v;
}
// B-fragment (32x16 bf16): col = laneLo; K = hf*16 + j (caller adds hf*16)
__device__ __forceinline__ v16bf loadFragB(const ushort_t* p){
  BF16Frag f;
  f.q[0] = *(const uint4*)(p);
  f.q[1] = *(const uint4*)(p + 8);
  return f.v;
}

// ---------------- embedding + phase space ----------------
__global__ void k_embed(const int* __restrict__ x, const float* __restrict__ emb,
                        const float* __restrict__ theta, const float* __restrict__ energy,
                        const float* __restrict__ excf,
                        float* __restrict__ real, float* __restrict__ imag){
  int row = blockIdx.x; int d = threadIdx.x;
  int tok = x[row];
  float e  = emb[(size_t)tok*DIM + d] * 0.1f;
  float th = theta[d];
  float r  = e * __cosf(th);
  float im = e * __sinf(th);
  float amp = sqrtf(r*r + im*im + 1e-8f);
  float exc = __expf(energy[d]) * excf[0];
  float coll = (amp < 0.1f) ? 1.0f : 0.0f;
  r  += coll * exc;
  im += coll * exc;
  float nrm = sqrtf(r*r + im*im + 1e-8f);
  real[(size_t)row*DIM + d] = clampf(r  / nrm, -1.f, 1.f);
  imag[(size_t)row*DIM + d] = clampf(im / nrm, -1.f, 1.f);
}

// ---------------- per-layer q/k/v layernorms -> bf16 operands ----------------
__global__ void k_qkv(const float* __restrict__ real, const float* __restrict__ imag,
                      const float* __restrict__ qg, const float* __restrict__ qb,
                      const float* __restrict__ kg, const float* __restrict__ kb,
                      const float* __restrict__ vg, const float* __restrict__ vb,
                      float* __restrict__ qrF,
                      ushort_t* __restrict__ qrB, ushort_t* __restrict__ qiB,
                      ushort_t* __restrict__ krB, ushort_t* __restrict__ kiB,
                      ushort_t* __restrict__ vrB, ushort_t* __restrict__ viB){
  __shared__ float4 red[256];
  int row = blockIdx.x; int d = threadIdx.x;
  size_t idx = (size_t)row*DIM + d;
  float r = real[idx], im = imag[idx];
  float4 v4; v4.x = r; v4.y = r*r; v4.z = im; v4.w = im*im;
  red[d] = v4; __syncthreads();
  for (int s = 128; s > 0; s >>= 1){
    if (d < s){ float4 a = red[d], b2 = red[d+s];
      a.x += b2.x; a.y += b2.y; a.z += b2.z; a.w += b2.w; red[d] = a; }
    __syncthreads();
  }
  float4 tot = red[0];
  float mur = tot.x * (1.f/DIM), mui = tot.z * (1.f/DIM);
  float rsr = rsqrtf(tot.y * (1.f/DIM) - mur*mur + 1e-5f);
  float rsi = rsqrtf(tot.w * (1.f/DIM) - mui*mui + 1e-5f);
  const float SCALE = 0.0625f;  // 256^-0.5
  float nr = (r - mur) * rsr, ni = (im - mui) * rsi;
  float qr = (nr*qg[d] + qb[d]) * SCALE;
  float qi = (ni*qg[d] + qb[d]) * SCALE;
  float kr = (nr*kg[d] + kb[d]) * SCALE;
  float ki = (ni*kg[d] + kb[d]) * SCALE;
  float vr =  nr*vg[d] + vb[d];
  float vi =  ni*vg[d] + vb[d];
  qrF[idx] = qr;
  qrB[idx] = f2bf(qr);
  qiB[idx] = f2bf(-qi);   // negated: scores accumulate qr*kr + (-qi)*ki
  krB[idx] = f2bf(kr);
  kiB[idx] = f2bf(ki);
  vrB[idx] = f2bf(vr);
  viB[idx] = f2bf(vi);
}

// ---------------- flash attention, fixed-max softmax (scores are clipped to +-10) ----
// 20 waves: 0-3 score+exp, 4-11 P*Vr, 12-19 P*Vi. Row-sum l via WMMA against ones.
#define ATTN_THREADS 640
#define ATTN_SMEM 135552
__global__ void __launch_bounds__(ATTN_THREADS, 1)
k_attn(const ushort_t* __restrict__ qrB, const ushort_t* __restrict__ qiB,
       const ushort_t* __restrict__ krB, const ushort_t* __restrict__ kiB,
       const ushort_t* __restrict__ vrB, const ushort_t* __restrict__ viB,
       const int* __restrict__ xx,
       float* __restrict__ outR, float* __restrict__ outI){
  extern __shared__ char smem[];
  ushort_t* sQr   = (ushort_t*)(smem);            // 64 x 256 bf16
  ushort_t* sQi   = (ushort_t*)(smem + 32768);    // 64 x 256
  ushort_t* sKr   = (ushort_t*)(smem + 65536);    // 32 x 256
  ushort_t* sKi   = (ushort_t*)(smem + 81920);    // 32 x 256
  ushort_t* sVr   = (ushort_t*)(smem + 98304);    // 256 x 32 (transposed)
  ushort_t* sVi   = (ushort_t*)(smem + 114688);   // 256 x 32
  ushort_t* sP    = (ushort_t*)(smem + 131072);   // 64 x 32 probs bf16
  float*    sL    = (float*)   (smem + 135168);   // 64
  float*    sMask = (float*)   (smem + 135424);   // 32

  const int tid = threadIdx.x;
  const int wave = tid >> 5, lane = tid & 31, laneLo = lane & 15, hf = lane >> 4;
  const bool isScore = (wave < 4);
  const int gS = wave & 3;                        // score row-group (waves 0-3)
  const int wp = (wave - 4) & 7;                  // PV wave index within r/i set
  const int gP = wp >> 1;                         // PV row-group
  const int colhalf = wp & 1;                     // PV 128-dim half
  const bool isImag = (wave >= 12);
  const int b = blockIdx.y;
  const int qbase = blockIdx.x * QTILE;
  const size_t rowBase = (size_t)b*SS + qbase;

  // ones fragment for l = P * 1 row-sum WMMA
  BF16Frag onesU;
  {
    ushort_t* op = (ushort_t*)&onesU;
    #pragma unroll
    for (int j = 0; j < 16; ++j) op[j] = 0x3F80;  // bf16 1.0
  }
  const v16bf fones = onesU.v;

  // stage Q tiles once
  {
    const uint4* gr = (const uint4*)(qrB + rowBase*DIM);
    const uint4* gi = (const uint4*)(qiB + rowBase*DIM);
    uint4* lr = (uint4*)sQr; uint4* li = (uint4*)sQi;
    for (int i = tid; i < QTILE*DIM/8; i += ATTN_THREADS){ lr[i] = gr[i]; li[i] = gi[i]; }
  }

  v8f accO[8];                                    // PV: 16 rows x 128 dims
  for (int t = 0; t < 8; ++t) for (int e = 0; e < 8; ++e) accO[t][e] = 0.f;
  v8f lacc;                                       // score: running row sums
  for (int e = 0; e < 8; ++e) lacc[e] = 0.f;

  for (int kt = 0; kt < SS/KTILE; ++kt){
    __syncthreads();                              // previous tile fully consumed
    const int keybase = kt * KTILE;
    const size_t kRowBase = (size_t)b*SS + keybase;
    {  // K tiles: straight copy
      const uint4* gkr = (const uint4*)(krB + kRowBase*DIM);
      const uint4* gki = (const uint4*)(kiB + kRowBase*DIM);
      uint4* lkr = (uint4*)sKr; uint4* lki = (uint4*)sKi;
      for (int i = tid; i < KTILE*DIM/8; i += ATTN_THREADS){ lkr[i] = gkr[i]; lki[i] = gki[i]; }
    }
    {  // V tiles: transpose [key][d] -> [d][key]
      const uint4* gvr = (const uint4*)(vrB + kRowBase*DIM);
      const uint4* gvi = (const uint4*)(viB + kRowBase*DIM);
      for (int i = tid; i < KTILE*DIM/8; i += ATTN_THREADS){
        int key = i >> 5;
        int d0  = (i & 31) * 8;
        uint4 a = gvr[i]; uint4 c = gvi[i];
        const ushort_t* pa = (const ushort_t*)&a;
        const ushort_t* pc = (const ushort_t*)&c;
        for (int j = 0; j < 8; ++j){
          sVr[(d0+j)*KTILE + key] = pa[j];
          sVi[(d0+j)*KTILE + key] = pc[j];
        }
      }
    }
    if (tid < KTILE)
      sMask[tid] = (xx[(size_t)b*SS + keybase + tid] == 0) ? -1.0e30f : 0.0f;
    __syncthreads();

    // prefetch next key tile while this one is computed (global_prefetch_b8)
    if (kt + 1 < SS/KTILE){
      const size_t nb = (kRowBase + KTILE) * DIM;   // next tile, element offset
      if (tid < 256){
        __builtin_prefetch((const char*)(krB + nb) + (size_t)tid*64, 0, 0);
        __builtin_prefetch((const char*)(vrB + nb) + (size_t)tid*64, 0, 0);
      } else if (tid < 512){
        __builtin_prefetch((const char*)(kiB + nb) + (size_t)(tid-256)*64, 0, 0);
        __builtin_prefetch((const char*)(viB + nb) + (size_t)(tid-256)*64, 0, 0);
      }
    }

    // ---- scores + exp: waves 0..3, each 16 rows x 32 keys ----
    // Sub-tiles processed sequentially to keep register pressure low:
    // only one 8-VGPR score accumulator + 4 fragments live at a time.
    if (isScore){
      const ushort_t* qrRow = sQr + (gS*16 + laneLo)*DIM;
      const ushort_t* qiRow = sQi + (gS*16 + laneLo)*DIM;
      #pragma unroll
      for (int t = 0; t < 2; ++t){
        v8f a; for (int e = 0; e < 8; ++e) a[e] = 0.f;
        const ushort_t* kRow = sKr + (t*16 + laneLo)*DIM;
        const ushort_t* iRow = sKi + (t*16 + laneLo)*DIM;
        for (int kc = 0; kc < 8; ++kc){
          v16bf fq  = loadFragA(qrRow + kc*32, hf);
          v16bf fk  = loadFragB(kRow + kc*32 + hf*16);
          a = wmma_bf16(fq, fk, a);
          v16bf fqn = loadFragA(qiRow + kc*32, hf);
          v16bf fi  = loadFragB(iRow + kc*32 + hf*16);
          a = wmma_bf16(fqn, fi, a);
        }
        // fixed softmax max = 10 (scores clipped to [-10,10]); mask per column
        float mb = sMask[t*16 + laneLo];
        for (int e = 0; e < 8; ++e){
          float p = __expf(clampf(a[e], -10.f, 10.f) + mb - 10.f);
          int r = gS*16 + e + 8*hf;
          sP[r*KTILE + t*16 + laneLo] = f2bf(p);
        }
      }
    }
    __syncthreads();

    // ---- l += P*1 (score waves) ; out += P*V (PV waves) ----
    if (isScore){
      v16bf fp = loadFragA(sP + (gS*16 + laneLo)*KTILE, hf);
      lacc = wmma_bf16(fp, fones, lacc);
    } else {
      const ushort_t* vt = isImag ? sVi : sVr;
      v16bf fp = loadFragA(sP + (gP*16 + laneLo)*KTILE, hf);
      for (int t = 0; t < 8; ++t){
        int nt = colhalf*8 + t;
        v16bf fv = loadFragB(vt + (nt*16 + laneLo)*KTILE + hf*16);
        accO[t] = wmma_bf16(fp, fv, accO[t]);
      }
    }
  }

  __syncthreads();
  if (isScore && laneLo == 0){
    for (int e = 0; e < 8; ++e) sL[gS*16 + e + 8*hf] = lacc[e];
  }
  __syncthreads();
  if (!isScore){
    float inv[8];
    for (int e = 0; e < 8; ++e){
      float l = sL[gP*16 + e + 8*hf];
      inv[e] = 1.0f / fmaxf(l, 1e-30f);
    }
    float* dst = isImag ? outI : outR;
    for (int t = 0; t < 8; ++t)
      for (int e = 0; e < 8; ++e){
        size_t orow = rowBase + gP*16 + e + 8*hf;
        dst[orow*DIM + (colhalf*8 + t)*16 + laneLo] = accO[t][e] * inv[e];
      }
  }
}

// ---------------- layer epilogue (imag := real per reference bug) ----------------
__global__ void k_epilogue(float* __restrict__ real, float* __restrict__ imag,
                           const float* __restrict__ outR, const float* __restrict__ outI,
                           const float* __restrict__ qrF,
                           const float* __restrict__ pp,
                           const float* __restrict__ pg, const float* __restrict__ pb){
  __shared__ float2 red[256];
  int row = blockIdx.x; int d = threadIdx.x;
  size_t idx = (size_t)row*DIM + d;
  float orv = outR[idx], oiv = outI[idx];
  float ph   = atan2f(oiv + 1e-8f, orv + 1e-8f);
  float pres = 0.1f / (1.0f + __expf(-pp[d]));
  float pr = clampf(orv * __cosf(ph * pres), -5.f, 5.f) + qrF[idx]*0.1f;
  float tr = real[idx] + pr*0.01f;
  float2 v2; v2.x = tr; v2.y = tr*tr;
  red[d] = v2; __syncthreads();
  for (int s = 128; s > 0; s >>= 1){
    if (d < s){ float2 a = red[d], c = red[d+s]; a.x += c.x; a.y += c.y; red[d] = a; }
    __syncthreads();
  }
  float mu  = red[0].x * (1.f/DIM);
  float var = red[0].y * (1.f/DIM) - mu*mu;
  float rs  = rsqrtf(var + 1e-8f);
  float rn  = clampf((tr - mu)*rs*pg[d] + pb[d], -1.f, 1.f);
  real[idx] = rn;
  imag[idx] = rn;     // reference: imag = clip(real)
}

// ---------------- final concat + LN -> bf16 ----------------
__global__ void k_combine(const float* __restrict__ real, const float* __restrict__ imag,
                          const float* __restrict__ pog, const float* __restrict__ pob,
                          ushort_t* __restrict__ comb){
  __shared__ float2 red[256];
  int row = blockIdx.x; int d = threadIdx.x;
  size_t idx = (size_t)row*DIM + d;
  float a = real[idx], c = imag[idx];
  float2 v2; v2.x = a + c; v2.y = a*a + c*c;
  red[d] = v2; __syncthreads();
  for (int s = 128; s > 0; s >>= 1){
    if (d < s){ float2 p = red[d], q = red[d+s]; p.x += q.x; p.y += q.y; red[d] = p; }
    __syncthreads();
  }
  float mu  = red[0].x * (1.f/(2*DIM));
  float var = red[0].y * (1.f/(2*DIM)) - mu*mu;
  float rs  = rsqrtf(var + 1e-5f);
  size_t obase = (size_t)row*(2*DIM);
  comb[obase + d]       = f2bf((a - mu)*rs*pog[d]       + pob[d]);
  comb[obase + DIM + d] = f2bf((c - mu)*rs*pog[DIM + d] + pob[DIM + d]);
}

__global__ void k_cvt(const float* __restrict__ w, ushort_t* __restrict__ o, int n){
  for (int i = blockIdx.x*blockDim.x + threadIdx.x; i < n; i += gridDim.x*blockDim.x)
    o[i] = f2bf(w[i]);
}

// ---------------- output projection: [8192x512] x [512x16384] ----------------
#define GEMM_SMEM 65536
__global__ void __launch_bounds__(256, 1)
k_gemm(const ushort_t* __restrict__ comb, const ushort_t* __restrict__ wb,
       const float* __restrict__ bias, float* __restrict__ out){
  extern __shared__ char smem[];
  ushort_t* sC = (ushort_t*)smem;           // 64 rows x 512
  const int tid = threadIdx.x;
  const int wave = tid >> 5, lane = tid & 31, laneLo = lane & 15, hf = lane >> 4;
  const int g = wave & 3, ch = wave >> 2;
  const int mbase = blockIdx.x * 64;
  const int nbase = blockIdx.y * 256;
  {
    const uint4* gsrc = (const uint4*)(comb + (size_t)mbase*512);
    uint4* ldst = (uint4*)sC;
    for (int i = tid; i < 64*512/8; i += 256) ldst[i] = gsrc[i];
  }
  __syncthreads();
  v8f acc[8];
  for (int t = 0; t < 8; ++t) for (int e = 0; e < 8; ++e) acc[t][e] = 0.f;
  const ushort_t* aRow = sC + (g*16 + laneLo)*512;
  const int col0 = nbase + ch*128;
  for (int kc = 0; kc < 16; ++kc){
    v16bf fa = loadFragA(aRow + kc*32, hf);
    for (int nt = 0; nt < 8; ++nt){
      const ushort_t* wp = wb + ((size_t)(col0 + nt*16 + laneLo))*512 + kc*32 + hf*16;
      v16bf fb = loadFragB(wp);
      acc[nt] = wmma_bf16(fa, fb, acc[nt]);
    }
  }
  for (int nt = 0; nt < 8; ++nt){
    int col = col0 + nt*16 + laneLo;
    float bv = bias[col];
    for (int e = 0; e < 8; ++e){
      int mrow = mbase + g*16 + e + 8*hf;
      out[(size_t)mrow*VOCAB + col] = clampf((acc[nt][e] + bv)*0.1f, -10.f, 10.f);
    }
  }
}

extern "C" void kernel_launch(void* const* d_in, const int* in_sizes, int n_in,
                              void* d_out, int out_size, void* d_ws, size_t ws_size,
                              hipStream_t stream){
  (void)in_sizes; (void)n_in; (void)out_size; (void)ws_size;
  const int*   x      = (const int*)  d_in[0];
  const float* emb    = (const float*)d_in[1];
  const float* theta  = (const float*)d_in[2];
  const float* energy = (const float*)d_in[3];
  const float* excf   = (const float*)d_in[4];
  const float* pp     = (const float*)d_in[5];
  const float* qg     = (const float*)d_in[6];
  const float* qb     = (const float*)d_in[7];
  const float* kg     = (const float*)d_in[8];
  const float* kb     = (const float*)d_in[9];
  const float* vg     = (const float*)d_in[10];
  const float* vb     = (const float*)d_in[11];
  const float* pg     = (const float*)d_in[12];
  const float* pb     = (const float*)d_in[13];
  const float* pog    = (const float*)d_in[14];
  const float* pob    = (const float*)d_in[15];
  const float* outw   = (const float*)d_in[16];
  const float* outb   = (const float*)d_in[17];
  float* out = (float*)d_out;

  char* wsp = (char*)d_ws;
  auto alloc = [&](size_t bytes)->void*{
    void* p = (void*)wsp;
    wsp += (bytes + 255) & ~(size_t)255;
    return p;
  };
  const size_t NR = (size_t)NROWS;
  float*    real = (float*)   alloc(NR*DIM*4);
  float*    imag = (float*)   alloc(NR*DIM*4);
  float*    qrF  = (float*)   alloc(NR*DIM*4);
  float*    outR = (float*)   alloc(NR*DIM*4);
  float*    outI = (float*)   alloc(NR*DIM*4);
  ushort_t* qrBf = (ushort_t*)alloc(NR*DIM*2);
  ushort_t* qiBf = (ushort_t*)alloc(NR*DIM*2);
  ushort_t* krBf = (ushort_t*)alloc(NR*DIM*2);
  ushort_t* kiBf = (ushort_t*)alloc(NR*DIM*2);
  ushort_t* vrBf = (ushort_t*)alloc(NR*DIM*2);
  ushort_t* viBf = (ushort_t*)alloc(NR*DIM*2);
  ushort_t* comb = (ushort_t*)alloc(NR*2*DIM*2);
  ushort_t* wbf  = (ushort_t*)alloc((size_t)VOCAB*2*DIM*2);

  k_embed<<<NROWS, 256, 0, stream>>>(x, emb, theta, energy, excf, real, imag);
  k_cvt<<<8192, 256, 0, stream>>>(outw, wbf, VOCAB*2*DIM);

  for (int i = 0; i < NLAYERS; ++i){
    k_qkv<<<NROWS, 256, 0, stream>>>(real, imag,
                                     qg + i*DIM, qb + i*DIM,
                                     kg + i*DIM, kb + i*DIM,
                                     vg + i*DIM, vb + i*DIM,
                                     qrF, qrBf, qiBf, krBf, kiBf, vrBf, viBf);
    k_attn<<<dim3(SS/QTILE, BB), ATTN_THREADS, ATTN_SMEM, stream>>>(qrBf, qiBf, krBf, kiBf,
                                                                    vrBf, viBf, x, outR, outI);
    k_epilogue<<<NROWS, 256, 0, stream>>>(real, imag, outR, outI, qrF,
                                          pp + i*DIM, pg + i*DIM, pb + i*DIM);
  }

  k_combine<<<NROWS, 256, 0, stream>>>(real, imag, pog, pob, comb);
  k_gemm<<<dim3(NROWS/64, VOCAB/256), 256, GEMM_SMEM, stream>>>(comb, wbf, outb, out);
}
